// RoPEAttention_21500606284465
// MI455X (gfx1250) — matrix operand order repaired and checked by reading, not compiled, not run on hardware
//
#include <hip/hip_runtime.h>
#include <hip/hip_bf16.h>

typedef __attribute__((ext_vector_type(16))) _Float16 v16h;
typedef __attribute__((ext_vector_type(8)))  float    v8f;

namespace {
constexpr int kB   = 16;
constexpr int kN   = 1025;
constexpr int kC   = 768;
constexpr int kH   = 12;
constexpr int kHD  = 64;
constexpr int kM   = kB * kN;     // 16400 flat tokens
constexpr int kNPad = 1056;       // multiple of 32 >= kN
constexpr int kBH  = kB * kH;     // 192

// f16 workspace layout (element offsets). attn output aliases XF (x is dead by then).
constexpr size_t E_X   = (size_t)kM * kC;
constexpr size_t E_WQ  = (size_t)3 * kC * kC;
constexpr size_t E_WP  = (size_t)kC * kC;
constexpr size_t E_QKV = (size_t)kBH * kNPad * kHD;
constexpr size_t OFF_X  = 0;
constexpr size_t OFF_WQ = OFF_X + E_X;
constexpr size_t OFF_WP = OFF_WQ + E_WQ;
constexpr size_t OFF_Q  = OFF_WP + E_WP;
constexpr size_t OFF_K  = OFF_Q + E_QKV;
constexpr size_t OFF_V  = OFF_K + E_QKV;
}

union F8 { v8f v; float f[8]; };

__device__ inline v16h load_frag(const _Float16* p0, const _Float16* p1) {
  union { v16h h; uint4 q[2]; } u;
  u.q[0] = *reinterpret_cast<const uint4*>(p0);
  u.q[1] = *reinterpret_cast<const uint4*>(p1);
  return u.h;
}

__device__ inline v8f wmma_f16(v16h a, v16h b, v8f c) {
  return __builtin_amdgcn_wmma_f32_16x16x32_f16(false, a, false, b, (short)0, c, false, false);
}

// CDNA5 async global->LDS copy (16B per lane), tracked by ASYNCcnt.
__device__ inline void async_ld_b128(unsigned int lds_off, const void* gp) {
  asm volatile("global_load_async_to_lds_b128 %0, %1, off"
               :: "v"(lds_off), "v"(gp)
               : "memory");
}
__device__ inline void wait_async0() {
  asm volatile("s_wait_asynccnt 0x0" ::: "memory");
}
__device__ inline unsigned int lds_off32(const void* p) {
  // low 32 bits of a flat shared address are the LDS byte offset (ISA 10.2)
  return (unsigned int)(uintptr_t)p;
}

// ---------------- kernel 0a: f32 -> f16 convert ----------------
__global__ void k_convert(const float* __restrict__ s, _Float16* __restrict__ d, int n) {
  int i = blockIdx.x * blockDim.x + threadIdx.x;
  int stride = gridDim.x * blockDim.x;
  for (; i < n; i += stride) d[i] = (_Float16)s[i];
}

// ---------------- kernel 0b: zero padded token rows of q/k/v^T ----------------
__global__ void k_zero_pads(_Float16* qf, _Float16* kf, _Float16* vtf) {
  constexpr int PADN = kNPad - kN;                // 31
  int idx = blockIdx.x * blockDim.x + threadIdx.x;
  int total = kBH * PADN * kHD;
  if (idx >= total) return;
  int d  = idx % kHD;
  int t  = idx / kHD;
  int np = t % PADN;
  int bh = t / PADN;
  int n  = kN + np;
  qf [((size_t)bh * kNPad + n) * kHD + d] = (_Float16)0.f;
  kf [((size_t)bh * kNPad + n) * kHD + d] = (_Float16)0.f;
  vtf[((size_t)bh * kHD  + d) * kNPad + n] = (_Float16)0.f;
}

// ---------------- kernel 1: QKV GEMM + bias + RoPE + scale + scatter ----------------
// grid (ceil(M/128)=129, 3C/64=36), block 128 (4 waves). Each wave: 32M x 64N tile, K=768.
__global__ __launch_bounds__(128) void k_qkv(
    const _Float16* __restrict__ xf, const _Float16* __restrict__ wq,
    const float* __restrict__ bqkv, const float* __restrict__ rope,
    _Float16* __restrict__ qf, _Float16* __restrict__ kf, _Float16* __restrict__ vtf)
{
  const int lane = threadIdx.x & 31;
  const int wave = threadIdx.x >> 5;
  const int col  = lane & 15;
  const int half = lane >> 4;
  const int m_base = blockIdx.x * 128 + wave * 32;
  if (m_base >= kM) return;
  const int o_base = blockIdx.y * 64;

  F8 acc[2][4];
#pragma unroll
  for (int g = 0; g < 2; ++g)
#pragma unroll
    for (int a = 0; a < 4; ++a)
#pragma unroll
      for (int r = 0; r < 8; ++r) acc[g][a].f[r] = 0.f;

  const _Float16* xrow0 = xf + (size_t)(m_base + col) * kC;
  const _Float16* xrow1 = xf + (size_t)(m_base + 16 + col) * kC;
  const int aoff = half * 8;
  const int boff = half * 8;
  for (int k0 = 0; k0 < kC; k0 += 32) {
    v16h af0 = load_frag(xrow0 + k0 + aoff, xrow0 + k0 + aoff + 16);
    v16h af1 = load_frag(xrow1 + k0 + aoff, xrow1 + k0 + aoff + 16);
#pragma unroll
    for (int a = 0; a < 4; ++a) {
      const _Float16* wrow = wq + (size_t)(o_base + a * 16 + col) * kC + k0 + boff;
      v16h bfr = load_frag(wrow, wrow + 16);
      acc[0][a].v = wmma_f16(af0, bfr, acc[0][a].v);
      acc[1][a].v = wmma_f16(af1, bfr, acc[1][a].v);
    }
  }

  const int which = o_base / kC;          // 0=q 1=k 2=v (tile never straddles)
  const int h     = (o_base % kC) / kHD;  // head
  float bias[4];
#pragma unroll
  for (int a = 0; a < 4; ++a) bias[a] = bqkv[o_base + a * 16 + col];

#pragma unroll
  for (int g = 0; g < 2; ++g) {
#pragma unroll
    for (int r = 0; r < 8; ++r) {
      int m = m_base + g * 16 + r + half * 8;
      if (m >= kM) continue;
      int b = m / kN;
      int n = m % kN;
      float t[4];
#pragma unroll
      for (int a = 0; a < 4; ++a) t[a] = acc[g][a].f[r] + bias[a];
      if (which == 2) {
#pragma unroll
        for (int a = 0; a < 4; ++a) {
          int d = a * 16 + col;
          vtf[((size_t)(b * kH + h) * kHD + d) * kNPad + n] = (_Float16)t[a];
        }
      } else {
        float o4[4];
        if (n >= 1) {  // RoPE; partner d±32 lives in acc[g][a±2] in the same lane
          const float* rp = rope + (size_t)(n - 1) * (2 * kHD);
#pragma unroll
          for (int a = 0; a < 4; ++a) {
            int d = a * 16 + col;
            float sn = rp[d];
            float cs = rp[kHD + d];
            float rot = (a < 2) ? -t[a + 2] : t[a - 2];
            o4[a] = t[a] * cs + rot * sn;
          }
        } else {
#pragma unroll
          for (int a = 0; a < 4; ++a) o4[a] = t[a];
        }
        if (which == 0) {
#pragma unroll
          for (int a = 0; a < 4; ++a) o4[a] *= 0.125f;  // hd^-0.5 folded into q
        }
        _Float16* dst = (which == 0) ? qf : kf;
        size_t rowbase = ((size_t)(b * kH + h) * kNPad + n) * kHD;
#pragma unroll
        for (int a = 0; a < 4; ++a) dst[rowbase + a * 16 + col] = (_Float16)o4[a];
      }
    }
  }
}

// ---------------- kernel 2: flash attention ----------------
// Block = 4 waves = 64 query rows of one (b,h); k/v tiles staged cooperatively into LDS
// with global_load_async_to_lds_b128. grid (ceil(65/4)=17, 192), block 128.
__global__ __launch_bounds__(128) void k_attn(
    const _Float16* __restrict__ qf, const _Float16* __restrict__ kfm,
    const _Float16* __restrict__ vtf, _Float16* __restrict__ attno)
{
  __shared__ _Float16 kS[32 * 64];        // [key][d]   4KB
  __shared__ _Float16 vS[64 * 32];        // [d][key]   4KB
  __shared__ _Float16 Plds[4][16 * 32];   // per-wave P staging, 4KB
  const int tid  = threadIdx.x;
  const int lane = tid & 31;
  const int wave = tid >> 5;
  const int col  = lane & 15;
  const int half = lane >> 4;
  const int bh = blockIdx.y;
  const int qb = blockIdx.x * 64 + wave * 16;

  const _Float16* qbase = qf  + (size_t)bh * kNPad * kHD;
  const _Float16* kbase = kfm + (size_t)bh * kNPad * kHD;
  const _Float16* vbase = vtf + (size_t)bh * kHD * kNPad;

  int qrow_idx = qb + col;
  if (qrow_idx >= kNPad) qrow_idx = 0;   // over-grid waves compute garbage, never store
  const _Float16* qrow = qbase + (size_t)qrow_idx * kHD;
  v16h qfr0 = load_frag(qrow + half * 8,      qrow + half * 8 + 16);
  v16h qfr1 = load_frag(qrow + 32 + half * 8, qrow + 32 + half * 8 + 16);

  float mrow[8], lrow[8];
  F8 acc[4];
#pragma unroll
  for (int r = 0; r < 8; ++r) { mrow[r] = -1e30f; lrow[r] = 0.f; }
#pragma unroll
  for (int a = 0; a < 4; ++a)
#pragma unroll
    for (int r = 0; r < 8; ++r) acc[a].f[r] = 0.f;

  _Float16* Pw = Plds[wave];

  for (int jb = 0; jb < kN; jb += 32) {
    // ---- cooperative async stage of k (32x64) and v (64x32) tiles ----
#pragma unroll
    for (int c = tid; c < 256; c += 128) {
      int kr = c >> 3, ke = (c & 7) * 8;   // k: 8 chunks of 8 f16 per 64-wide row
      async_ld_b128(lds_off32(&kS[kr * 64 + ke]),
                    kbase + (size_t)(jb + kr) * kHD + ke);
      int vr = c >> 2, ve = (c & 3) * 8;   // v: 4 chunks per 32-wide row
      async_ld_b128(lds_off32(&vS[vr * 32 + ve]),
                    vbase + (size_t)vr * kNPad + jb + ve);
    }
    wait_async0();
    __syncthreads();

    // ---- S = q @ k^T (B-frags from LDS) ----
    F8 S[2];
#pragma unroll
    for (int t = 0; t < 2; ++t) {
      const _Float16* krow = kS + (t * 16 + col) * 64;
      v16h kb0 = load_frag(krow + half * 8,      krow + half * 8 + 16);
      v16h kb1 = load_frag(krow + 32 + half * 8, krow + 32 + half * 8 + 16);
      v8f s = {0.f, 0.f, 0.f, 0.f, 0.f, 0.f, 0.f, 0.f};
      s = wmma_f16(qfr0, kb0, s);
      s = wmma_f16(qfr1, kb1, s);
      S[t].v = s;
      if (jb + t * 16 + col >= kN) {
#pragma unroll
        for (int r = 0; r < 8; ++r) S[t].f[r] = -1e30f;
      }
    }

    // ---- online softmax row stats (16-lane xor reductions stay in row-halves) ----
    float alpha[8];
#pragma unroll
    for (int r = 0; r < 8; ++r) {
      float mx = fmaxf(S[0].f[r], S[1].f[r]);
#pragma unroll
      for (int msk = 1; msk <= 8; msk <<= 1) mx = fmaxf(mx, __shfl_xor(mx, msk, 32));
      float mnew = fmaxf(mrow[r], mx);
      float p0 = __expf(S[0].f[r] - mnew);
      float p1 = __expf(S[1].f[r] - mnew);
      S[0].f[r] = p0; S[1].f[r] = p1;
      float rs = p0 + p1;
#pragma unroll
      for (int msk = 1; msk <= 8; msk <<= 1) rs += __shfl_xor(rs, msk, 32);
      alpha[r] = __expf(mrow[r] - mnew);
      lrow[r] = alpha[r] * lrow[r] + rs;
      mrow[r] = mnew;
    }

    // ---- stage P (D-layout) -> LDS, preload v B-frags ----
#pragma unroll
    for (int r = 0; r < 8; ++r) {
      Pw[(r + half * 8) * 32 + col]      = (_Float16)S[0].f[r];
      Pw[(r + half * 8) * 32 + 16 + col] = (_Float16)S[1].f[r];
    }
    v16h vb[4];
#pragma unroll
    for (int a = 0; a < 4; ++a) {
      const _Float16* vrow = vS + (a * 16 + col) * 32 + half * 8;
      vb[a] = load_frag(vrow, vrow + 16);
    }
    __syncthreads();
    const _Float16* prow = Pw + col * 32;
    v16h pfr = load_frag(prow + half * 8, prow + half * 8 + 16);

    // ---- rescale accumulators, add P @ V ----
#pragma unroll
    for (int a = 0; a < 4; ++a) {
      F8 c;
#pragma unroll
      for (int r = 0; r < 8; ++r) c.f[r] = acc[a].f[r] * alpha[r];
      acc[a].v = wmma_f16(pfr, vb[a], c.v);
    }
    __syncthreads();  // all LDS reads done before next iteration's async writes
  }

  const int b = bh / kH;
  const int h = bh % kH;
#pragma unroll
  for (int r = 0; r < 8; ++r) {
    int row = qb + r + half * 8;
    if (row >= kN) continue;
    float inv = 1.0f / lrow[r];
#pragma unroll
    for (int a = 0; a < 4; ++a) {
      attno[((size_t)(b * kN + row)) * kC + h * kHD + a * 16 + col] =
          (_Float16)(acc[a].f[r] * inv);
    }
  }
}

// ---------------- kernel 3: output projection (fp32 out + bias) ----------------
// grid (129, 12), block 128 (4 waves), wave tile 32M x 64N, K=768.
__global__ __launch_bounds__(128) void k_proj(
    const _Float16* __restrict__ af, const _Float16* __restrict__ wp,
    const float* __restrict__ bp, float* __restrict__ out)
{
  const int lane = threadIdx.x & 31;
  const int wave = threadIdx.x >> 5;
  const int col  = lane & 15;
  const int half = lane >> 4;
  const int m_base = blockIdx.x * 128 + wave * 32;
  if (m_base >= kM) return;
  const int o_base = blockIdx.y * 64;

  F8 acc[2][4];
#pragma unroll
  for (int g = 0; g < 2; ++g)
#pragma unroll
    for (int a = 0; a < 4; ++a)
#pragma unroll
      for (int r = 0; r < 8; ++r) acc[g][a].f[r] = 0.f;

  const _Float16* arow0 = af + (size_t)(m_base + col) * kC;
  const _Float16* arow1 = af + (size_t)(m_base + 16 + col) * kC;
  const int aoff = half * 8;
  const int boff = half * 8;
  for (int k0 = 0; k0 < kC; k0 += 32) {
    v16h af0 = load_frag(arow0 + k0 + aoff, arow0 + k0 + aoff + 16);
    v16h af1 = load_frag(arow1 + k0 + aoff, arow1 + k0 + aoff + 16);
#pragma unroll
    for (int a = 0; a < 4; ++a) {
      const _Float16* wrow = wp + (size_t)(o_base + a * 16 + col) * kC + k0 + boff;
      v16h bfr = load_frag(wrow, wrow + 16);
      acc[0][a].v = wmma_f16(af0, bfr, acc[0][a].v);
      acc[1][a].v = wmma_f16(af1, bfr, acc[1][a].v);
    }
  }

  float bias[4];
#pragma unroll
  for (int a = 0; a < 4; ++a) bias[a] = bp[o_base + a * 16 + col];
#pragma unroll
  for (int g = 0; g < 2; ++g)
#pragma unroll
    for (int r = 0; r < 8; ++r) {
      int m = m_base + g * 16 + r + half * 8;
      if (m >= kM) continue;
#pragma unroll
      for (int a = 0; a < 4; ++a)
        out[(size_t)m * kC + o_base + a * 16 + col] = acc[g][a].f[r] + bias[a];
    }
}

extern "C" void kernel_launch(void* const* d_in, const int* in_sizes, int n_in,
                              void* d_out, int out_size, void* d_ws, size_t ws_size,
                              hipStream_t stream) {
  (void)in_sizes; (void)n_in; (void)out_size; (void)ws_size;
  const float* x     = (const float*)d_in[0];
  const float* rope  = (const float*)d_in[1];
  const float* Wqkv  = (const float*)d_in[2];
  const float* bqkv  = (const float*)d_in[3];
  const float* Wproj = (const float*)d_in[4];
  const float* bproj = (const float*)d_in[5];
  float* out = (float*)d_out;

  _Float16* wsh = (_Float16*)d_ws;
  _Float16* XF  = wsh + OFF_X;   // also reused as attention output (x dead after k_qkv)
  _Float16* WQ  = wsh + OFF_WQ;
  _Float16* WP  = wsh + OFF_WP;
  _Float16* QF  = wsh + OFF_Q;
  _Float16* KF  = wsh + OFF_K;
  _Float16* VT  = wsh + OFF_V;
  _Float16* AT  = XF;

  k_convert<<<4096, 256, 0, stream>>>(x,     XF, (int)E_X);
  k_convert<<<2048, 256, 0, stream>>>(Wqkv,  WQ, (int)E_WQ);
  k_convert<<<1024, 256, 0, stream>>>(Wproj, WP, (int)E_WP);

  {
    int total = kBH * (kNPad - kN) * kHD;
    k_zero_pads<<<(total + 255) / 256, 256, 0, stream>>>(QF, KF, VT);
  }

  k_qkv<<<dim3((kM + 127) / 128, (3 * kC) / 64), 128, 0, stream>>>(
      XF, WQ, bqkv, rope, QF, KF, VT);

  {
    int qblocks = ((kN + 15) / 16 + 3) / 4;  // 17
    k_attn<<<dim3(qblocks, kBH), 128, 0, stream>>>(QF, KF, VT, AT);
  }

  k_proj<<<dim3((kM + 127) / 128, kC / 64), 128, 0, stream>>>(AT, WP, bproj, out);
}
